// MKGED_3530463117950
// MI455X (gfx1250) — matrix operand order repaired and checked by reading, not compiled, not run on hardware
//
#include <hip/hip_runtime.h>
#include <hip/hip_bf16.h>
#include <math.h>

// ---------------------------------------------------------------- constants
#define VSZ   15000
#define RSZ   237
#define DD    768
#define HH    768
#define BB    64
#define NNP1  40            // NN+1
#define NS    10240         // B*4*(NN+1) samples
#define G4    3072          // 4*H gate width
#define KC    1536          // D + H concat K
#define FF    4608          // 3*2*H
#define NGRP  256           // NS / 40
#define NPATH 128           // B*2

typedef __attribute__((ext_vector_type(16))) __bf16 v16bf;
typedef __attribute__((ext_vector_type(8)))  float  v8f;

union BFrag { uint4 q[2]; v16bf v; };

static __device__ __forceinline__ __bf16 f2bf(float f) {
  unsigned u = __builtin_bit_cast(unsigned, f);
  unsigned r = u + 0x7FFFu + ((u >> 16) & 1u);
  unsigned short s = (unsigned short)(r >> 16);
  return __builtin_bit_cast(__bf16, s);
}
static __device__ __forceinline__ float sigm(float x) { return 1.0f / (1.0f + expf(-x)); }

// ---------------------------------------------------------------- max(entity_scores)
__global__ void reduce_max_kernel(const float* __restrict__ es, int n, float* __restrict__ out) {
  __shared__ float sm[256];
  float m = -1e30f;
  for (int i = threadIdx.x; i < n; i += 256) m = fmaxf(m, es[i]);
  sm[threadIdx.x] = m; __syncthreads();
  for (int s = 128; s > 0; s >>= 1) {
    if (threadIdx.x < s) sm[threadIdx.x] = fmaxf(sm[threadIdx.x], sm[threadIdx.x + s]);
    __syncthreads();
  }
  if (threadIdx.x == 0) *out = sm[0];
}

// ---------------------------------------------------------------- gather/aggregate -> bf16 X [NS,3,D]
__global__ void aggregate_kernel(const int* __restrict__ hrt,
                                 const float* __restrict__ text, const float* __restrict__ img,
                                 const float* __restrict__ rel,  const float* __restrict__ es,
                                 const float* __restrict__ maxp, __bf16* __restrict__ X) {
  int s = blockIdx.x;
  int h = hrt[3 * s], r = hrt[3 * s + 1], t = hrt[3 * s + 2];
  float ms = *maxp;
  float hw = (ms - es[h]) / ms, tw = (ms - es[t]) / ms;
  const float* th = text + (size_t)h * DD; const float* ih = img + (size_t)h * DD;
  const float* tt = text + (size_t)t * DD; const float* it = img + (size_t)t * DD;
  const float* rr = rel  + (size_t)r * DD;
  __bf16* xs = X + (size_t)s * (3 * DD);
  for (int d = threadIdx.x; d < DD; d += 256) {
    xs[d]          = f2bf(th[d] + hw * ih[d]);
    xs[DD + d]     = f2bf(rr[d]);
    xs[2 * DD + d] = f2bf(tt[d] + tw * it[d]);
  }
}

// ---------------------------------------------------------------- weight prep
__global__ void convert_cat_kernel(const float* __restrict__ wih, const float* __restrict__ whh,
                                   __bf16* __restrict__ wcat) {
  size_t i = (size_t)blockIdx.x * 256 + threadIdx.x;
  if (i >= (size_t)G4 * KC) return;
  int n = (int)(i / KC), k = (int)(i % KC);
  float v = (k < DD) ? wih[(size_t)n * DD + k] : whh[(size_t)n * HH + (k - DD)];
  wcat[i] = f2bf(v);
}

__global__ void transpose_bf16_kernel(const float* __restrict__ Wg, __bf16* __restrict__ Wt) {
  size_t i = (size_t)blockIdx.x * 256 + threadIdx.x;
  if (i >= (size_t)FF * FF) return;
  int n = (int)(i / FF), k = (int)(i % FF);
  Wt[i] = f2bf(Wg[(size_t)k * FF + n]);   // Wt[n][k] = W_gat[k][n]
}

// ---------------------------------------------------------------- WMMA bf16 GEMM
// C[M,N] = A[M,K] * W[N,K]^T ; A,W bf16 row-major, C fp32.
// block = 256 thr (8 waves), block tile 64(M) x 128(N), wave tile 16x64.
__global__ void __launch_bounds__(256) gemm_bf16_wmma(
    const __bf16* __restrict__ A, int lda,
    const __bf16* __restrict__ W, int ldw,
    float* __restrict__ C, int ldc, int K) {
  const int lane = threadIdx.x & 31;
  const int wave = threadIdx.x >> 5;
  const int wr = wave & 3;
  const int wc = wave >> 2;
  const int m0 = blockIdx.y * 64 + wr * 16;
  const int n0 = blockIdx.x * 128 + wc * 64;
  const int l15 = lane & 15;
  const int hi  = lane >> 4;

  const __bf16* arow  = A + (size_t)(m0 + l15) * lda + hi * 16;   // A frag: k in [koff,koff+16)
  const __bf16* wrow0 = W + (size_t)(n0 +  0 + l15) * ldw + hi * 16;
  const __bf16* wrow1 = W + (size_t)(n0 + 16 + l15) * ldw + hi * 16;
  const __bf16* wrow2 = W + (size_t)(n0 + 32 + l15) * ldw + hi * 16;
  const __bf16* wrow3 = W + (size_t)(n0 + 48 + l15) * ldw + hi * 16;

  v8f acc0 = {}, acc1 = {}, acc2 = {}, acc3 = {};

  for (int k0 = 0; k0 < K; k0 += 32) {
    BFrag fa;
    fa.q[0] = *reinterpret_cast<const uint4*>(arow + k0);
    fa.q[1] = *reinterpret_cast<const uint4*>(arow + k0 + 8);
    BFrag f0, f1, f2, f3;
    f0.q[0] = *reinterpret_cast<const uint4*>(wrow0 + k0);
    f0.q[1] = *reinterpret_cast<const uint4*>(wrow0 + k0 + 8);
    f1.q[0] = *reinterpret_cast<const uint4*>(wrow1 + k0);
    f1.q[1] = *reinterpret_cast<const uint4*>(wrow1 + k0 + 8);
    f2.q[0] = *reinterpret_cast<const uint4*>(wrow2 + k0);
    f2.q[1] = *reinterpret_cast<const uint4*>(wrow2 + k0 + 8);
    f3.q[0] = *reinterpret_cast<const uint4*>(wrow3 + k0);
    f3.q[1] = *reinterpret_cast<const uint4*>(wrow3 + k0 + 8);
    acc0 = __builtin_amdgcn_wmma_f32_16x16x32_bf16(false, fa.v, false, f0.v, (short)0, acc0, false, false);
    acc1 = __builtin_amdgcn_wmma_f32_16x16x32_bf16(false, fa.v, false, f1.v, (short)0, acc1, false, false);
    acc2 = __builtin_amdgcn_wmma_f32_16x16x32_bf16(false, fa.v, false, f2.v, (short)0, acc2, false, false);
    acc3 = __builtin_amdgcn_wmma_f32_16x16x32_bf16(false, fa.v, false, f3.v, (short)0, acc3, false, false);
  }

  const int mst = m0 + hi * 8;
  for (int r = 0; r < 8; r++) {
    float* crow = C + (size_t)(mst + r) * ldc + n0 + l15;
    crow[0]  = acc0[r];
    crow[16] = acc1[r];
    crow[32] = acc2[r];
    crow[48] = acc3[r];
  }
}

// ---------------------------------------------------------------- LSTM step helpers
// Abuf[s, 0:768] = X[s, t_actual, :]; first step also zeros h-part and c.
__global__ void lstm_pack_kernel(const __bf16* __restrict__ X, __bf16* __restrict__ Abuf,
                                 float* __restrict__ cbuf, int t_actual, int first) {
  size_t idx = (size_t)blockIdx.x * 256 + threadIdx.x;   // NS*DD
  int s = (int)(idx / DD), j = (int)(idx % DD);
  Abuf[(size_t)s * KC + j] = X[(size_t)s * (3 * DD) + (size_t)t_actual * DD + j];
  if (first) {
    Abuf[(size_t)s * KC + DD + j] = f2bf(0.0f);
    cbuf[idx] = 0.0f;
  }
}

// gates -> c,h ; writes h (bf16) into Abuf h-slot and packed GAT input.
__global__ void lstm_gates_kernel(const float* __restrict__ G,
                                  const float* __restrict__ bih, const float* __restrict__ bhh,
                                  float* __restrict__ cbuf, __bf16* __restrict__ Abuf,
                                  __bf16* __restrict__ outb, int t_actual, int dir) {
  size_t idx = (size_t)blockIdx.x * 256 + threadIdx.x;   // NS*HH
  int s = (int)(idx / HH), j = (int)(idx % HH);
  const float* g = G + (size_t)s * G4;
  float gi = g[j]            + bih[j]            + bhh[j];
  float gf = g[HH + j]       + bih[HH + j]       + bhh[HH + j];
  float gg = g[2 * HH + j]   + bih[2 * HH + j]   + bhh[2 * HH + j];
  float go = g[3 * HH + j]   + bih[3 * HH + j]   + bhh[3 * HH + j];
  float c = sigm(gf) * cbuf[idx] + sigm(gi) * tanhf(gg);
  float h = sigm(go) * tanhf(c);
  cbuf[idx] = c;
  Abuf[(size_t)s * KC + DD + j] = f2bf(h);
  outb[(size_t)s * FF + (size_t)t_actual * (2 * HH) + (size_t)dir * HH + j] = f2bf(h);
}

// ---------------------------------------------------------------- ls_pos / ls_neg
__global__ void ls_kernel(const int* __restrict__ hrt, const float* __restrict__ text,
                          const float* __restrict__ img, const float* __restrict__ rel,
                          const float* __restrict__ es, const float* __restrict__ maxp,
                          float* __restrict__ lsp, float* __restrict__ lsn) {
  int b = blockIdx.x;
  __shared__ float red[256];
  float ms = *maxp;
  for (int which = 0; which < 2; which++) {
    int s = b * 160 + which * 80;
    int h = hrt[3 * s], r = hrt[3 * s + 1], t = hrt[3 * s + 2];
    float hw = (ms - es[h]) / ms, tw = (ms - es[t]) / ms;
    float acc = 0.0f;
    for (int d = threadIdx.x; d < DD; d += 256) {
      float hd = text[(size_t)h * DD + d] + hw * img[(size_t)h * DD + d];
      float td = text[(size_t)t * DD + d] + tw * img[(size_t)t * DD + d];
      float v = hd + rel[(size_t)r * DD + d] - td;
      acc += v * v;
    }
    red[threadIdx.x] = acc; __syncthreads();
    for (int st = 128; st > 0; st >>= 1) {
      if (threadIdx.x < st) red[threadIdx.x] += red[threadIdx.x + st];
      __syncthreads();
    }
    if (threadIdx.x == 0) { if (which) lsn[b] = sqrtf(red[0]); else lsp[b] = sqrtf(red[0]); }
    __syncthreads();
  }
}

// ---------------------------------------------------------------- u = h.a1 , v = h.a2
__global__ void uv_kernel(const float* __restrict__ hbuf, const float* __restrict__ ag,
                          float* __restrict__ u, float* __restrict__ v) {
  int s = blockIdx.x;
  __shared__ float ru[256], rv[256];
  float au = 0.0f, av = 0.0f;
  const float* hr = hbuf + (size_t)s * FF;
  for (int d = threadIdx.x; d < FF; d += 256) {
    float hv = hr[d];
    au += hv * ag[d];
    av += hv * ag[FF + d];
  }
  ru[threadIdx.x] = au; rv[threadIdx.x] = av; __syncthreads();
  for (int st = 128; st > 0; st >>= 1) {
    if (threadIdx.x < st) { ru[threadIdx.x] += ru[threadIdx.x + st]; rv[threadIdx.x] += rv[threadIdx.x + st]; }
    __syncthreads();
  }
  if (threadIdx.x == 0) { u[s] = ru[0]; v[s] = rv[0]; }
}

// ---------------------------------------------------------------- attention + weighted sum
__global__ void att_kernel(const float* __restrict__ hbuf, const float* __restrict__ u,
                           const float* __restrict__ v, float* __restrict__ oatt) {
  int g = blockIdx.x;   // 0..255
  __shared__ float att[NNP1];
  if (threadIdx.x < NNP1) {
    float e = u[g * NNP1 + threadIdx.x] + v[g * NNP1];
    att[threadIdx.x] = (e >= 0.0f) ? e : 0.05f * e;   // leaky_relu 0.05
  }
  __syncthreads();
  if (threadIdx.x == 0) {
    float mx = -1e30f;
    for (int j = 0; j < NNP1; j++) mx = fmaxf(mx, att[j]);
    float sum = 0.0f;
    for (int j = 0; j < NNP1; j++) { att[j] = expf(att[j] - mx); sum += att[j]; }
    for (int j = 0; j < NNP1; j++) att[j] = fmaxf(att[j] / sum - 0.005f, 0.0f);
  }
  __syncthreads();
  for (int d = threadIdx.x; d < FF; d += 256) {
    float acc = 0.0f;
    for (int j = 0; j < NNP1; j++)
      acc += att[j] * hbuf[(size_t)(g * NNP1 + j) * FF + d];
    oatt[(size_t)g * FF + d] = acc;
  }
}

// ---------------------------------------------------------------- gs_pos / gs_neg
__global__ void gs_kernel(const float* __restrict__ oa, float* __restrict__ gsp, float* __restrict__ gsn) {
  int b = blockIdx.x;
  __shared__ float red[256];
  for (int which = 0; which < 2; which++) {
    const float* x = oa + (size_t)(b * 4 + which * 2) * FF;
    const float* y = oa + (size_t)(b * 4 + which * 2 + 1) * FF;
    float acc = 0.0f;
    for (int d = threadIdx.x; d < FF; d += 256) { float dv = x[d] - y[d]; acc += dv * dv; }
    red[threadIdx.x] = acc; __syncthreads();
    for (int st = 128; st > 0; st >>= 1) {
      if (threadIdx.x < st) red[threadIdx.x] += red[threadIdx.x + st];
      __syncthreads();
    }
    if (threadIdx.x == 0) { if (which) gsn[b] = sqrtf(red[0]); else gsp[b] = sqrtf(red[0]); }
    __syncthreads();
  }
}

// ---------------------------------------------------------------- path scoring -> g[b,i]
__global__ void path_kernel(const int* __restrict__ ap_all, const float* __restrict__ text,
                            const float* __restrict__ img, const float* __restrict__ rel,
                            const float* __restrict__ es, const float* __restrict__ maxp,
                            float* __restrict__ gout) {
  int bi = blockIdx.x;  // 0..127
  const int* ap = ap_all + bi * 4 * 3 * 3;  // [P][L][3]
  float ms = *maxp;
  __shared__ float red[256];
  __shared__ float s1[4], s2[4];
  for (int p = 0; p < 4; p++) {
    float accl0 = 0, accl1 = 0, accl2 = 0, acc2 = 0;
    for (int d = threadIdx.x; d < DD; d += 256) {
      float head0 = 0, tail2 = 0, relsum = 0;
      for (int l = 0; l < 3; l++) {
        int h = ap[(p * 3 + l) * 3 + 0];
        int r = ap[(p * 3 + l) * 3 + 1];
        int t = ap[(p * 3 + l) * 3 + 2];
        float hw = (ms - es[h]) / ms, tw = (ms - es[t]) / ms;
        float hd = text[(size_t)h * DD + d] + hw * img[(size_t)h * DD + d];
        float td = text[(size_t)t * DD + d] + tw * img[(size_t)t * DD + d];
        float rd = rel[(size_t)r * DD + d];
        float vv = hd + rd - td;
        if (l == 0) { accl0 += vv * vv; head0 = hd; }
        else if (l == 1) accl1 += vv * vv;
        else { accl2 += vv * vv; tail2 = td; }
        relsum += rd;
      }
      float v2 = head0 + relsum - tail2;
      acc2 += v2 * v2;
    }
    float vals[4] = {accl0, accl1, accl2, acc2};
    for (int q = 0; q < 4; q++) {
      red[threadIdx.x] = vals[q]; __syncthreads();
      for (int st = 128; st > 0; st >>= 1) {
        if (threadIdx.x < st) red[threadIdx.x] += red[threadIdx.x + st];
        __syncthreads();
      }
      if (threadIdx.x == 0) {
        if (q == 0) s1[p] = sqrtf(red[0]);
        else if (q < 3) s1[p] += sqrtf(red[0]);
        else s2[p] = sqrtf(red[0]);
      }
      __syncthreads();
    }
  }
  if (threadIdx.x == 0) {
    float sc[4], mx = -1e30f;
    for (int p = 0; p < 4; p++) { sc[p] = s1[p] / 3.0f; mx = fmaxf(mx, sc[p]); }
    float sum = 0, w[4];
    for (int p = 0; p < 4; p++) { w[p] = expf(sc[p] - mx); sum += w[p]; }
    float g = 0;
    for (int p = 0; p < 4; p++) g += (w[p] / sum) * s2[p];
    gout[bi] = g * 0.25f;   // mean over P
  }
}

// ---------------------------------------------------------------- final loss + pos_score
__global__ void final_kernel(const float* __restrict__ lsp, const float* __restrict__ lsn,
                             const float* __restrict__ gsp, const float* __restrict__ gsn,
                             const float* __restrict__ gbuf, const int* __restrict__ idx,
                             const float* __restrict__ soft, float* __restrict__ out) {
  __shared__ float sm[64];
  int b = threadIdx.x;
  float lsl = fmaxf(lsp[b] - lsn[b] + 1.0f, 0.0f);
  float gsl = fmaxf(gsp[b] - gsn[b] + 1.0f, 0.0f);
  float gpp = gbuf[b * 2], gpn = gbuf[b * 2 + 1];
  float gpl = fmaxf(gpp - gpn + 1.0f, 0.0f);
  out[1 + b] = lsp[b] + 1.0f * gsp[b] + 1.0f * gpp;   // LAM*10 = DELTA*10 = 1.0
  sm[b] = soft[idx[b]] * (lsl + gsl + gpl);
  __syncthreads();
  for (int st = 32; st > 0; st >>= 1) {
    if (b < st) sm[b] += sm[b + st];
    __syncthreads();
  }
  if (b == 0) out[0] = sm[0];
}

// ================================================================ launcher
extern "C" void kernel_launch(void* const* d_in, const int* in_sizes, int n_in,
                              void* d_out, int out_size, void* d_ws, size_t ws_size,
                              hipStream_t stream) {
  (void)in_sizes; (void)n_in; (void)out_size; (void)ws_size;
  const int*   hrt   = (const int*)d_in[0];
  const int*   apath = (const int*)d_in[1];
  const int*   idx   = (const int*)d_in[2];
  const float* soft  = (const float*)d_in[3];
  const float* text  = (const float*)d_in[4];
  const float* img   = (const float*)d_in[5];
  const float* rel   = (const float*)d_in[6];
  const float* es    = (const float*)d_in[7];
  const float* wihf  = (const float*)d_in[8];
  const float* whhf  = (const float*)d_in[9];
  const float* bihf  = (const float*)d_in[10];
  const float* bhhf  = (const float*)d_in[11];
  const float* wihb  = (const float*)d_in[12];
  const float* whhb  = (const float*)d_in[13];
  const float* bihb  = (const float*)d_in[14];
  const float* bhhb  = (const float*)d_in[15];
  const float* Wg    = (const float*)d_in[16];
  const float* ag    = (const float*)d_in[17];
  float* out = (float*)d_out;

  // workspace carve-out
  char* ws = (char*)d_ws;
  size_t off = 0;
  auto alloc = [&](size_t bytes) -> void* {
    void* p = ws + off;
    off += (bytes + 255) & ~(size_t)255;
    return p;
  };
  float*  d_max  = (float*) alloc(sizeof(float));
  __bf16* Xb     = (__bf16*)alloc((size_t)NS * 3 * DD * 2);
  __bf16* wcatf  = (__bf16*)alloc((size_t)G4 * KC * 2);
  __bf16* wcatb  = (__bf16*)alloc((size_t)G4 * KC * 2);
  __bf16* Wgt    = (__bf16*)alloc((size_t)FF * FF * 2);
  __bf16* Abuf   = (__bf16*)alloc((size_t)NS * KC * 2);
  float*  Gbuf   = (float*) alloc((size_t)NS * G4 * 4);
  float*  cbuf   = (float*) alloc((size_t)NS * HH * 4);
  __bf16* outb   = (__bf16*)alloc((size_t)NS * FF * 2);
  float*  hbuf   = (float*) alloc((size_t)NS * FF * 4);
  float*  ubuf   = (float*) alloc((size_t)NS * 4);
  float*  vbuf   = (float*) alloc((size_t)NS * 4);
  float*  oatt   = (float*) alloc((size_t)NGRP * FF * 4);
  float*  lsp    = (float*) alloc(BB * 4);
  float*  lsn    = (float*) alloc(BB * 4);
  float*  gsp    = (float*) alloc(BB * 4);
  float*  gsn    = (float*) alloc(BB * 4);
  float*  gbuf   = (float*) alloc(NPATH * 4);

  // 1. scalar prep
  reduce_max_kernel<<<1, 256, 0, stream>>>(es, VSZ, d_max);
  aggregate_kernel<<<NS, 256, 0, stream>>>(hrt, text, img, rel, es, d_max, Xb);
  {
    int total = G4 * KC;
    convert_cat_kernel<<<(total + 255) / 256, 256, 0, stream>>>(wihf, whhf, wcatf);
    convert_cat_kernel<<<(total + 255) / 256, 256, 0, stream>>>(wihb, whhb, wcatb);
  }
  {
    size_t total = (size_t)FF * FF;
    transpose_bf16_kernel<<<(unsigned)((total + 255) / 256), 256, 0, stream>>>(Wg, Wgt);
  }
  ls_kernel<<<BB, 256, 0, stream>>>(hrt, text, img, rel, es, d_max, lsp, lsn);

  // 2. bidirectional LSTM (fused gate GEMM: [NS,KC] x [G4,KC]^T with WMMA)
  const int ew_blocks = (NS * HH) / 256;    // 30720
  dim3 ggrid(G4 / 128, NS / 64);            // (24,160)
  for (int dir = 0; dir < 2; dir++) {
    const __bf16* wcat = dir ? wcatb : wcatf;
    const float* bih = dir ? bihb : bihf;
    const float* bhh = dir ? bhhb : bhhf;
    for (int tt = 0; tt < 3; tt++) {
      int t_actual = dir ? (2 - tt) : tt;
      lstm_pack_kernel<<<ew_blocks, 256, 0, stream>>>(Xb, Abuf, cbuf, t_actual, tt == 0);
      gemm_bf16_wmma<<<ggrid, 256, 0, stream>>>(Abuf, KC, wcat, KC, Gbuf, G4, KC);
      lstm_gates_kernel<<<ew_blocks, 256, 0, stream>>>(Gbuf, bih, bhh, cbuf, Abuf, outb,
                                                       t_actual, dir);
    }
  }

  // 3. GAT projection: h = out @ W_gat  ([NS,FF] x [FF,FF])
  dim3 hgrid(FF / 128, NS / 64);            // (36,160)
  gemm_bf16_wmma<<<hgrid, 256, 0, stream>>>(outb, FF, Wgt, FF, hbuf, FF, FF);

  // 4. attention + graph scores
  uv_kernel<<<NS, 256, 0, stream>>>(hbuf, ag, ubuf, vbuf);
  att_kernel<<<NGRP, 256, 0, stream>>>(hbuf, ubuf, vbuf, oatt);
  gs_kernel<<<BB, 256, 0, stream>>>(oatt, gsp, gsn);

  // 5. path scores
  path_kernel<<<NPATH, 256, 0, stream>>>(apath, text, img, rel, es, d_max, gbuf);

  // 6. final reduction
  final_kernel<<<1, 64, 0, stream>>>(lsp, lsn, gsp, gsn, gbuf, idx, soft, out);
}
